// MambaSegDecoder_37555194036768
// MI455X (gfx1250) — matrix operand order repaired and checked by reading, not compiled, not run on hardware
//
#include <hip/hip_runtime.h>
#include <hip/hip_bf16.h>
#include <stdint.h>
#include <stddef.h>

// ---------------------------------------------------------------- types
typedef __bf16 bf16_t;
typedef bf16_t v16bf __attribute__((ext_vector_type(16)));
typedef float  v8f   __attribute__((ext_vector_type(8)));

union FragAB { v16bf v; uint4 q[2]; };

// ---------------------------------------------------------------- helpers
__device__ __forceinline__ unsigned short f2bf(float f) {
  union { float f; unsigned int u; } c; c.f = f;
  unsigned int u = c.u;
  unsigned int r = (u + 0x7FFFu + ((u >> 16) & 1u)) >> 16;   // RNE
  return (unsigned short)r;
}
__device__ __forceinline__ float bf2f(unsigned short h) {
  union { unsigned int u; float f; } c; c.u = ((unsigned int)h) << 16; return c.f;
}
__device__ __forceinline__ float sigmoidf_(float x) { return 1.f / (1.f + __expf(-x)); }
__device__ __forceinline__ float siluf_(float x)    { return x * sigmoidf_(x); }

// scan-order position l  <->  spatial raster position (involutive per direction)
__device__ __forceinline__ int map_dir(int k, int l, int H, int W) {
  int L = H * W;
  if (k == 0) return l;
  if (k == 1) { int h = l % H, w = l / H; return h * W + w; }
  if (k == 2) return L - 1 - l;
  int lp = L - 1 - l; int h = lp % H, w = lp / H; return h * W + w;
}

// ---------------------------------------------------------------- GEMM (bf16 WMMA)
// C[M,Np](f32) = A[M,Kp](bf16 row-major) * Wt[Kp,Np](bf16 row-major) (+bias[n])
// Wave computes a 16(M) x NV*16(N) register block; NV is a COMPILE-TIME constant
// so the WMMA quad is straight-line (no EXEC predication around WMMA, fixed
// accumulator VGPRs, software-pipelined loads). A fragment is loaded once per
// K-step and reused by NV WMMAs. Wave->tile mapping is N-block-major so the 8
// waves of a block stream the same weight panel (panels live in the 192MB L2).
// A fragment: lane = M(sub), per-lane contiguous K chunks (ISA 16-bit A 16x32).
// B fragment: lane = N column, per-lane 16 K values in the same K order as A (ISA 16-bit B).
template <int NV>
__global__ void __launch_bounds__(256)
k_gemm_bf16(const unsigned short* __restrict__ A,
            const unsigned short* __restrict__ Wt,
            float* __restrict__ C, const float* __restrict__ bias,
            int M, int Kp, int Np)
{
  int wave = blockIdx.x * 8 + (threadIdx.x >> 5);
  int lane = threadIdx.x & 31;
  int mTiles = M >> 4;
  int mIdx = wave % mTiles;
  int nIdx = wave / mTiles;
  int tN = nIdx * (NV * 16);
  if (tN >= Np) return;
  int tM = mIdx << 4;

  int sub   = lane & 15;
  int khalf = (lane >> 4) << 3;

  const unsigned short* ap = A  + (size_t)(tM + sub) * Kp + khalf;
  const unsigned short* wp = Wt + tN + sub;                    // column tN+sub

  v8f acc[NV];
#pragma unroll
  for (int t = 0; t < NV; ++t) acc[t] = (v8f){0.f, 0.f, 0.f, 0.f, 0.f, 0.f, 0.f, 0.f};

  for (int k0 = 0; k0 < Kp; k0 += 32) {
    FragAB fa;
    fa.q[0] = *(const uint4*)(ap + k0);
    fa.q[1] = *(const uint4*)(ap + k0 + 16);
    const unsigned short* wr = wp + (size_t)k0 * Np;
    __builtin_prefetch(ap + k0 + 128, 0, 1);                   // global_prefetch_b8
    __builtin_prefetch(wr + (size_t)32 * Np, 0, 1);
#pragma unroll
    for (int t = 0; t < NV; ++t) {
      FragAB fb;
#pragma unroll
      for (int s = 0; s < 16; ++s) {
        int kk = (s < 8) ? (khalf + s) : (16 + khalf + (s - 8));
        ((unsigned short*)&fb)[s] = wr[(size_t)kk * Np + 16 * t];
      }
      acc[t] = __builtin_amdgcn_wmma_f32_16x16x32_bf16(
          false, fa.v, false, fb.v, (short)0, acc[t], false, false);
    }
  }

  int rbase = tM + ((lane >> 4) << 3);
#pragma unroll
  for (int t = 0; t < NV; ++t) {
    int cN = tN + 16 * t + sub;
    float bv = bias ? bias[cN] : 0.f;
#pragma unroll
    for (int i = 0; i < 8; ++i)
      C[(size_t)(rbase + i) * Np + cN] = acc[t][i] + bv;
  }
}

// ---------------------------------------------------------------- weight panel builder
// Wt[Kp,Np] bf16 from f32 source; srcKN=1: src is (K,N) row-major, else (N,K).
__global__ void k_build_wt(const float* __restrict__ W, unsigned short* __restrict__ Wt,
                           int K, int N, int Kp, int Np, int srcKN)
{
  int idx = blockIdx.x * blockDim.x + threadIdx.x;
  if (idx >= Kp * Np) return;
  int k = idx / Np, n = idx % Np;
  float v = 0.f;
  if (k < K && n < N) v = srcKN ? W[(size_t)k * N + n] : W[(size_t)n * K + k];
  Wt[idx] = f2bf(v);
}

// ---------------------------------------------------------------- elementwise kernels
__global__ void k_f32_to_bf16(const float* __restrict__ s, unsigned short* __restrict__ d, int n) {
  int i = blockIdx.x * blockDim.x + threadIdx.x;
  if (i < n) d[i] = f2bf(s[i]);
}
__global__ void k_fill0(float* __restrict__ p, int n) {
  int i = blockIdx.x * blockDim.x + threadIdx.x;
  if (i < n) p[i] = 0.f;
}
__global__ void k_add(float* __restrict__ a, const float* __restrict__ b, int n) {
  int i = blockIdx.x * blockDim.x + threadIdx.x;
  if (i < n) a[i] += b[i];
}

// wave-per-row LayerNorm (wave32 shuffle reduction); in-place safe.
__global__ void __launch_bounds__(256)
k_ln(const float* __restrict__ x, const float* __restrict__ g, const float* __restrict__ b,
     float* __restrict__ y, int M, int C)
{
  int row  = blockIdx.x * 8 + (threadIdx.x >> 5);
  int lane = threadIdx.x & 31;
  if (row >= M) return;
  const float* xr = x + (size_t)row * C;
  float s = 0.f, s2 = 0.f;
  for (int c = lane; c < C; c += 32) { float v = xr[c]; s += v; s2 += v * v; }
#pragma unroll
  for (int o = 16; o > 0; o >>= 1) { s += __shfl_xor(s, o, 32); s2 += __shfl_xor(s2, o, 32); }
  float invn = 1.f / (float)C;
  float mu   = s * invn;
  float var  = fmaxf(s2 * invn - mu * mu, 0.f);
  float inv  = rsqrtf(var + 1e-5f);
  float* yr = y + (size_t)row * C;
  for (int c = lane; c < C; c += 32) yr[c] = (xr[c] - mu) * inv * g[c] + b[c];
}

// NCHW f32 -> rows (b,h,w) x C bf16
__global__ void k_nchw_rows_bf(const float* __restrict__ src, unsigned short* __restrict__ dst,
                               int Bq, int C, int H, int W)
{
  int idx = blockIdx.x * blockDim.x + threadIdx.x;
  int tot = Bq * H * W * C;
  if (idx >= tot) return;
  int c = idx % C, row = idx / C;
  int l = row % (H * W), b = row / (H * W);
  int h = l / W, w = l % W;
  dst[idx] = f2bf(src[(((size_t)b * C + c) * H + h) * W + w]);
}

// 2x2 patch-expand rearrange: t rows (b,h,w) x 4C -> out (b,2h+r,2w+c2) x C
__global__ void k_patch_expand2(const float* __restrict__ t, float* __restrict__ out,
                                int Bq, int H, int W, int C)
{
  int idx = blockIdx.x * blockDim.x + threadIdx.x;
  int tot = Bq * 2 * H * 2 * W * C;
  if (idx >= tot) return;
  int ch = idx % C; int rest = idx / C;
  int ww = rest % (2 * W); rest /= (2 * W);
  int hh = rest % (2 * H); int b = rest / (2 * H);
  int h = hh >> 1, r = hh & 1, w = ww >> 1, c2 = ww & 1;
  out[idx] = t[((size_t)(b * H + h) * W + w) * (4 * C) + (size_t)(r * 2 + c2) * C + ch];
}

// concat [xexp rows | skip NCHW] -> bf16 rows x 2C
__global__ void k_concat_bf(const float* __restrict__ xexp, const float* __restrict__ skip,
                            unsigned short* __restrict__ dst, int Bq, int C, int H, int W)
{
  int idx = blockIdx.x * blockDim.x + threadIdx.x;
  int tot = Bq * H * W * 2 * C;
  if (idx >= tot) return;
  int col = idx % (2 * C), row = idx / (2 * C);
  int l = row % (H * W), b = row / (H * W);
  int h = l / W, w = l % W;
  float v;
  if (col < C) v = xexp[(size_t)row * C + col];
  else         v = skip[(((size_t)b * C + (col - C)) * H + h) * W + w];
  dst[idx] = f2bf(v);
}

// depthwise 3x3 SAME + SiLU; xin = in_proj output rows (ld = 2*di), first di cols
__global__ void k_dwconv_silu(const float* __restrict__ xin, const float* __restrict__ cw,
                              const float* __restrict__ cb, float* __restrict__ xc,
                              int Bq, int di, int H, int W, int ldin)
{
  int idx = blockIdx.x * blockDim.x + threadIdx.x;
  int tot = Bq * H * W * di;
  if (idx >= tot) return;
  int d = idx % di, row = idx / di;
  int l = row % (H * W), b = row / (H * W);
  int h = l / W, w = l % W;
  float acc = cb[d];
#pragma unroll
  for (int kh = 0; kh < 3; ++kh)
#pragma unroll
    for (int kw = 0; kw < 3; ++kw) {
      int hh = h + kh - 1, ww = w + kw - 1;
      if (hh >= 0 && hh < H && ww >= 0 && ww < W)
        acc += cw[(size_t)d * 9 + kh * 3 + kw] *
               xin[((size_t)b * H * W + (size_t)hh * W + ww) * ldin + d];
    }
  xc[idx] = siluf_(acc);
}

// gather direction-k activation (scan order) as bf16
__global__ void k_build_dir_bf(const float* __restrict__ xc, unsigned short* __restrict__ Xk,
                               int k, int Bq, int di, int H, int W)
{
  int L = H * W;
  int idx = blockIdx.x * blockDim.x + threadIdx.x;
  int tot = Bq * L * di;
  if (idx >= tot) return;
  int d = idx % di, row = idx / di;
  int l = row % L, b = row / L;
  int ls = map_dir(k, l, H, W);
  Xk[idx] = f2bf(xc[((size_t)b * L + ls) * di + d]);
}

// dts slice (cols 0..dr-1 of xd, ld 48) -> bf16 M x 32 (zero padded)
__global__ void k_dts_bf(const float* __restrict__ xd, unsigned short* __restrict__ dts,
                         int M, int dr)
{
  int idx = blockIdx.x * blockDim.x + threadIdx.x;
  if (idx >= M * 32) return;
  int j = idx % 32, m = idx / 32;
  float v = (j < dr) ? xd[(size_t)m * 48 + j] : 0.f;
  dts[idx] = f2bf(v);
}

__global__ void k_softplus(float* __restrict__ dt, const float* __restrict__ dtb, int M, int di)
{
  int idx = blockIdx.x * blockDim.x + threadIdx.x;
  if (idx >= M * di) return;
  int d = idx % di;
  float x = dt[idx] + dtb[d];
  dt[idx] = (x > 20.f) ? x : log1pf(__expf(x));
}

// selective scan, thread per (b,d) for one direction; 16 states in registers
__global__ void k_scan(const float* __restrict__ delta, const unsigned short* __restrict__ u,
                       const float* __restrict__ xd, const float* __restrict__ Alog,
                       const float* __restrict__ Ds, float* __restrict__ ys,
                       int Bq, int di, int L, int dr)
{
  int tid = blockIdx.x * blockDim.x + threadIdx.x;
  if (tid >= Bq * di) return;
  int b = tid / di, d = tid % di;
  float A[16];
#pragma unroll
  for (int n = 0; n < 16; ++n) A[n] = -__expf(Alog[(size_t)d * 16 + n]);
  float Dv = Ds[d];
  float h[16];
#pragma unroll
  for (int n = 0; n < 16; ++n) h[n] = 0.f;
  const float*          dl = delta + (size_t)b * L * di + d;
  const unsigned short* ul = u     + (size_t)b * L * di + d;
  const float*          bc = xd    + (size_t)b * L * 48;
  float*                yo = ys    + (size_t)b * L * di + d;
  for (int l = 0; l < L; ++l) {
    float dt = dl[(size_t)l * di];
    float uv = bf2f(ul[(size_t)l * di]);
    const float* row = bc + (size_t)l * 48;
    float du = dt * uv, y = 0.f;
#pragma unroll
    for (int n = 0; n < 16; ++n) {
      float dA = __expf(dt * A[n]);
      h[n] = dA * h[n] + du * row[dr + n];
      y += h[n] * row[dr + 16 + n];
    }
    yo[(size_t)l * di] = y + Dv * uv;
  }
}

// scatter-accumulate direction-k scan output back to spatial order
__global__ void k_accum_dir(const float* __restrict__ ys, float* __restrict__ y,
                            int k, int Bq, int di, int H, int W)
{
  int L = H * W;
  int idx = blockIdx.x * blockDim.x + threadIdx.x;
  int tot = Bq * L * di;
  if (idx >= tot) return;
  int d = idx % di, row = idx / di;
  int l = row % L, b = row / L;
  int ls = map_dir(k, l, H, W);                // bijection per k -> no atomics needed
  y[((size_t)b * L + ls) * di + d] += ys[idx];
}

// g = bf16( yln * silu(z) ), z = inproj[:, di + d]
__global__ void k_gate_bf(const float* __restrict__ yln, const float* __restrict__ inproj,
                          unsigned short* __restrict__ g, int M, int di)
{
  int idx = blockIdx.x * blockDim.x + threadIdx.x;
  if (idx >= M * di) return;
  int d = idx % di, m = idx / di;
  float z = inproj[(size_t)m * 2 * di + di + d];
  g[idx] = f2bf(yln[idx] * siluf_(z));
}

// seg finalize: sgo rows m=((b*Hc+h0)*Wc+w0)*16+g, g=r*4+c2 -> NCHW out + bias
__global__ void k_seg_out(const float* __restrict__ sgo, const float* __restrict__ segb,
                          float* __restrict__ out, int Bq, int Hc, int Wc)
{
  int idx = blockIdx.x * blockDim.x + threadIdx.x;
  int tot = Bq * Hc * Wc * 16 * 6;
  if (idx >= tot) return;
  int o = idx % 6, m = idx / 6;
  int g = m % 16; int mm = m / 16;
  int w0 = mm % Wc; mm /= Wc;
  int h0 = mm % Hc; int b = mm / Hc;
  int r = g >> 2, c2 = g & 3;
  int h = h0 * 4 + r, w = w0 * 4 + c2;
  int Hh = Hc * 4, Ww = Wc * 4;
  out[(((size_t)b * 6 + o) * Hh + h) * Ww + w] = sgo[(size_t)m * 16 + o] + segb[o];
}

// ---------------------------------------------------------------- host side
namespace {
struct Bump {
  char* base; size_t off;
  void* take(size_t bytes) {
    size_t a = (off + 255) & ~(size_t)255;
    off = a + bytes;
    return base + a;
  }
};
static inline unsigned blks(size_t n) { return (unsigned)((n + 255) / 256); }
}

extern "C" void kernel_launch(void* const* d_in, const int* in_sizes, int n_in,
                              void* d_out, int out_size, void* d_ws, size_t ws_size,
                              hipStream_t stream)
{
  (void)in_sizes; (void)n_in; (void)out_size; (void)ws_size;
  const int Bq = 4;
  const int enc[4] = {64, 128, 256, 512};

  auto in = [&](int i) { return (const float*)d_in[i]; };
  const float* skips[4] = { in(0), in(1), in(2), in(3) };
  // params flattened in sorted-key pytree order after the 4 skips:
  const float* fexp_b = in(4);
  const float* fexp_g = in(5);
  const float* fexp_w = in(6);
  const float* seg_b  = in(7);
  const float* seg_w  = in(8);
  // stage s base = 9 + s*31; block j base = stageBase + j*13
  // block leaves: 0 A_log, 1 Ds, 2 conv_b, 3 conv_w, 4 dt_b, 5 dt_w, 6 in_w,
  //               7 ln_b, 8 ln_g, 9 on_b, 10 on_g, 11 out_w, 12 xproj_w
  // stage extras: +26 cbd_b, +27 cbd_w, +28 exp_b, +29 exp_g, +30 exp_w

  Bump ws{ (char*)d_ws, 0 };
  float* xping = (float*)ws.take((size_t)16384 * 64 * sizeof(float));  // 4 MB
  float* xpong = (float*)ws.take((size_t)16384 * 64 * sizeof(float));  // 4 MB
  size_t stageMark = ws.off;

  // NV specialization: every Np in this net is a multiple of 64, exactly 48,
  // or exactly 16 -> per-launch compile-time subtile count, no divergence.
  auto gemm = [&](const unsigned short* A, const unsigned short* Wt, float* C,
                  const float* bias, int M, int Kp, int Np) {
    int mTiles = M / 16;
    if (Np % 64 == 0) {
      int waves = mTiles * (Np / 64);
      k_gemm_bf16<4><<<(waves + 7) / 8, 256, 0, stream>>>(A, Wt, C, bias, M, Kp, Np);
    } else if (Np == 48) {
      k_gemm_bf16<3><<<(mTiles + 7) / 8, 256, 0, stream>>>(A, Wt, C, bias, M, Kp, Np);
    } else { // Np == 16
      k_gemm_bf16<1><<<(mTiles + 7) / 8, 256, 0, stream>>>(A, Wt, C, bias, M, Kp, Np);
    }
  };
  auto buildWt = [&](const float* W, int K, int N, int Kp, int Np, bool srcKN) {
    unsigned short* Wt = (unsigned short*)ws.take((size_t)Kp * Np * 2);
    k_build_wt<<<blks((size_t)Kp * Np), 256, 0, stream>>>(W, Wt, K, N, Kp, Np, srcKN ? 1 : 0);
    return Wt;
  };
  auto ln = [&](const float* x, const float* g, const float* b, float* y, int M, int C) {
    k_ln<<<(unsigned)((M + 7) / 8), 256, 0, stream>>>(x, g, b, y, M, C);
  };

  float* xcur = nullptr;

  for (int s = 0; s < 3; ++s) {
    ws.off = stageMark;
    int din = enc[3 - s], dskip = enc[2 - s];
    int Hin = 8 << s, Win = 8 << s;
    int H = Hin * 2, W = Win * 2;
    int Mprev = Bq * Hin * Win;
    int L = H * W, M = Bq * L;
    int d = dskip, di = 2 * d, dr = (d + 15) / 16;
    int base = 9 + s * 31;
    const float* cbd_b = in(base + 26);
    const float* cbd_w = in(base + 27);
    const float* exp_b = in(base + 28);
    const float* exp_g = in(base + 29);
    const float* exp_w = in(base + 30);

    // ---- patch expand: GEMM (Mprev x din) x (din x 2din), 2x2 rearrange, LN
    unsigned short* Aexp = (unsigned short*)ws.take((size_t)Mprev * din * 2);
    if (s == 0)
      k_nchw_rows_bf<<<blks((size_t)Mprev * din), 256, 0, stream>>>(skips[3], Aexp, Bq, din, Hin, Win);
    else
      k_f32_to_bf16<<<blks((size_t)Mprev * din), 256, 0, stream>>>(xcur, Aexp, Mprev * din);
    unsigned short* Wexp = buildWt(exp_w, din, 2 * din, din, 2 * din, true);
    float* texp = (float*)ws.take((size_t)Mprev * 2 * din * 4);
    gemm(Aexp, Wexp, texp, nullptr, Mprev, din, 2 * din);
    int C2 = din / 2;                       // == dskip
    float* xexp = (float*)ws.take((size_t)M * C2 * 4);
    k_patch_expand2<<<blks((size_t)M * C2), 256, 0, stream>>>(texp, xexp, Bq, Hin, Win, C2);
    ln(xexp, exp_g, exp_b, xexp, M, C2);

    // ---- concat with skip + cbd projection (+bias)
    unsigned short* Acat = (unsigned short*)ws.take((size_t)M * 2 * dskip * 2);
    k_concat_bf<<<blks((size_t)M * 2 * dskip), 256, 0, stream>>>(xexp, skips[2 - s], Acat, Bq, dskip, H, W);
    unsigned short* Wcbd = buildWt(cbd_w, 2 * dskip, dskip, 2 * dskip, dskip, true);
    float* xnew = (s % 2 == 0) ? xping : xpong;
    gemm(Acat, Wcbd, xnew, cbd_b, M, 2 * dskip, dskip);
    xcur = xnew;

    // ---- 2 VSS blocks
    size_t blockMark = ws.off;
    for (int j = 0; j < 2; ++j) {
      ws.off = blockMark;
      int bb = base + j * 13;
      const float* A_log  = in(bb + 0);
      const float* Ds_p   = in(bb + 1);
      const float* conv_b = in(bb + 2);
      const float* conv_w = in(bb + 3);
      const float* dt_b   = in(bb + 4);
      const float* dt_w   = in(bb + 5);
      const float* in_w   = in(bb + 6);
      const float* ln_b   = in(bb + 7);
      const float* ln_g   = in(bb + 8);
      const float* on_b   = in(bb + 9);
      const float* on_g   = in(bb + 10);
      const float* out_w  = in(bb + 11);
      const float* xprj_w = in(bb + 12);

      float* ln1 = (float*)ws.take((size_t)M * d * 4);
      ln(xcur, ln_g, ln_b, ln1, M, d);
      unsigned short* ln1b = (unsigned short*)ws.take((size_t)M * d * 2);
      k_f32_to_bf16<<<blks((size_t)M * d), 256, 0, stream>>>(ln1, ln1b, M * d);

      unsigned short* Win_ = buildWt(in_w, d, 2 * di, d, 2 * di, true);
      float* inproj = (float*)ws.take((size_t)M * 2 * di * 4);
      gemm(ln1b, Win_, inproj, nullptr, M, d, 2 * di);

      float* xc = (float*)ws.take((size_t)M * di * 4);
      k_dwconv_silu<<<blks((size_t)M * di), 256, 0, stream>>>(inproj, conv_w, conv_b, xc,
                                                              Bq, di, H, W, 2 * di);
      float* ycomb = (float*)ws.take((size_t)M * di * 4);
      k_fill0<<<blks((size_t)M * di), 256, 0, stream>>>(ycomb, M * di);

      size_t kMark = ws.off;
      for (int k = 0; k < 4; ++k) {
        ws.off = kMark;
        unsigned short* Xk = (unsigned short*)ws.take((size_t)M * di * 2);
        k_build_dir_bf<<<blks((size_t)M * di), 256, 0, stream>>>(xc, Xk, k, Bq, di, H, W);

        // x_proj: (c x di) stored (N,K); pad N -> 48
        unsigned short* Wxp = buildWt(xprj_w + (size_t)k * (dr + 32) * di,
                                      di, dr + 32, di, 48, false);
        float* xdk = (float*)ws.take((size_t)M * 48 * 4);
        gemm(Xk, Wxp, xdk, nullptr, M, di, 48);

        unsigned short* dtsb = (unsigned short*)ws.take((size_t)M * 32 * 2);
        k_dts_bf<<<blks((size_t)M * 32), 256, 0, stream>>>(xdk, dtsb, M, dr);

        // dt proj: (di x dr) stored (N,K); pad K -> 32
        unsigned short* Wdt = buildWt(dt_w + (size_t)k * di * dr, dr, di, 32, di, false);
        float* delta = (float*)ws.take((size_t)M * di * 4);
        gemm(dtsb, Wdt, delta, nullptr, M, 32, di);
        k_softplus<<<blks((size_t)M * di), 256, 0, stream>>>(delta, dt_b + (size_t)k * di, M, di);

        float* ysk = (float*)ws.take((size_t)M * di * 4);
        int nscan = Bq * di;
        k_scan<<<(unsigned)((nscan + 127) / 128), 128, 0, stream>>>(
            delta, Xk, xdk, A_log + (size_t)k * di * 16, Ds_p + (size_t)k * di,
            ysk, Bq, di, L, dr);
        k_accum_dir<<<blks((size_t)M * di), 256, 0, stream>>>(ysk, ycomb, k, Bq, di, H, W);
      }
      ws.off = kMark;

      ln(ycomb, on_g, on_b, ycomb, M, di);
      unsigned short* gbuf = (unsigned short*)ws.take((size_t)M * di * 2);
      k_gate_bf<<<blks((size_t)M * di), 256, 0, stream>>>(ycomb, inproj, gbuf, M, di);

      unsigned short* Wout = buildWt(out_w, di, d, di, d, true);
      float* outt = (float*)ws.take((size_t)M * d * 4);
      gemm(gbuf, Wout, outt, nullptr, M, di, d);
      k_add<<<blks((size_t)M * d), 256, 0, stream>>>(xcur, outt, M * d);
    }
  }

  // ---- final 4x expand + LN + seg head
  ws.off = stageMark;
  int Mf = Bq * 64 * 64;                      // 16384 rows of 64 ch (stage-2 output)
  unsigned short* Af = (unsigned short*)ws.take((size_t)Mf * 64 * 2);
  k_f32_to_bf16<<<blks((size_t)Mf * 64), 256, 0, stream>>>(xcur, Af, Mf * 64);
  unsigned short* Wf = buildWt(fexp_w, 64, 1024, 64, 1024, true);
  float* tf = (float*)ws.take((size_t)Mf * 1024 * 4);
  gemm(Af, Wf, tf, nullptr, Mf, 64, 1024);

  // LN over each 64-channel group (16 groups per row are contiguous slices)
  ln(tf, fexp_g, fexp_b, tf, Mf * 16, 64);

  unsigned short* tb = (unsigned short*)ws.take((size_t)Mf * 1024 * 2);
  k_f32_to_bf16<<<blks((size_t)Mf * 1024), 256, 0, stream>>>(tf, tb, Mf * 1024);

  unsigned short* Wseg = buildWt(seg_w, 64, 6, 64, 16, false);  // (6,64) is (N,K)
  float* sgo = (float*)ws.take((size_t)Mf * 16 * 16 * 4);
  gemm(tb, Wseg, sgo, nullptr, Mf * 16, 64, 16);

  k_seg_out<<<blks((size_t)Mf * 16 * 6), 256, 0, stream>>>(sgo, seg_b, (float*)d_out, Bq, 64, 64);
}
